// TinyGPT_26645977105198
// MI455X (gfx1250) — matrix operand order repaired and checked by reading, not compiled, not run on hardware
//
#include <hip/hip_runtime.h>
#include <cstdint>
#include <cstddef>

// ---------------- model constants ----------------
#define D_EMB   768
#define T_SEQ   1024
#define N_HEAD  12
#define N_LAYER 12
#define B_BATCH 2
#define BT      (B_BATCH * T_SEQ)   // 2048
#define F4DIM   (4 * D_EMB)         // 3072
#define D3      (3 * D_EMB)         // 2304
#define DHEAD   64
#define VOCAB   50257

// ---------------- wmma helper types ----------------
typedef __bf16 bf16;
typedef __bf16 v16bf __attribute__((ext_vector_type(16)));
typedef __bf16 v8bf  __attribute__((ext_vector_type(8)));
typedef __bf16 v4bf  __attribute__((ext_vector_type(4)));
typedef float  v8f   __attribute__((ext_vector_type(8)));

__device__ __forceinline__ bf16 f2bf(float f) {
  union { float f; uint32_t u; } c; c.f = f;
  uint32_t r = c.u + 0x7FFFu + ((c.u >> 16) & 1u);   // round-to-nearest-even
  uint16_t h = (uint16_t)(r >> 16);
  return __builtin_bit_cast(bf16, h);
}

__device__ __forceinline__ v4bf f2bf4(float4 v) {
  v4bf r;
  r[0] = f2bf(v.x); r[1] = f2bf(v.y); r[2] = f2bf(v.z); r[3] = f2bf(v.w);
  return r;
}

__device__ __forceinline__ v16bf cat16(v8bf lo, v8bf hi) {
  return __builtin_shufflevector(lo, hi, 0,1,2,3,4,5,6,7,8,9,10,11,12,13,14,15);
}

__device__ __forceinline__ v8f zero8() {
  v8f z;
#pragma unroll
  for (int i = 0; i < 8; ++i) z[i] = 0.f;
  return z;
}

__device__ __forceinline__ v8f wmma_bf16(v16bf a, v16bf b, v8f c) {
  // D = A(16x32 bf16) * B(32x16 bf16) + C(16x16 f32)
  return __builtin_amdgcn_wmma_f32_16x16x32_bf16(false, a, false, b, (short)0, c,
                                                 false, false);
}

// ---------------- embedding gather ----------------
__global__ __launch_bounds__(256)
void embed_kernel(const int* __restrict__ idx, const float* __restrict__ tok,
                  const float* __restrict__ pos, float* __restrict__ x) {
  const int row = blockIdx.x;             // b*T + t
  const int t   = row & (T_SEQ - 1);
  const int token = idx[row];
  const int tid = threadIdx.x;
#pragma unroll
  for (int i = 0; i < 3; ++i) {
    int d = tid + 256 * i;
    x[(size_t)row * D_EMB + d] =
        tok[(size_t)token * D_EMB + d] + pos[(size_t)t * D_EMB + d];
  }
}

// ---------------- layernorm (one block per row, D=768) ----------------
__global__ __launch_bounds__(256)
void ln_kernel(const float* __restrict__ x, const float* __restrict__ w,
               const float* __restrict__ b, float* __restrict__ o) {
  __shared__ float s1[256], s2[256];
  const int tid = threadIdx.x;
  const int row = blockIdx.x;
  const float* xr = x + (size_t)row * D_EMB;
  float v[3], a0 = 0.f, a1 = 0.f;
#pragma unroll
  for (int i = 0; i < 3; ++i) {
    v[i] = xr[tid + 256 * i];
    a0 += v[i];
    a1 += v[i] * v[i];
  }
  s1[tid] = a0; s2[tid] = a1;
  __syncthreads();
  for (int st = 128; st > 0; st >>= 1) {
    if (tid < st) { s1[tid] += s1[tid + st]; s2[tid] += s2[tid + st]; }
    __syncthreads();
  }
  const float mean = s1[0] * (1.f / D_EMB);
  const float var  = s2[0] * (1.f / D_EMB) - mean * mean;
  const float rstd = rsqrtf(var + 1e-5f);
  float* orow = o + (size_t)row * D_EMB;
#pragma unroll
  for (int i = 0; i < 3; ++i) {
    int d = tid + 256 * i;
    orow[d] = (v[i] - mean) * rstd * w[d] + b[d];
  }
}

// ---------------- WMMA GEMM: C[M,N] = act[M,K] @ W[K,N] (+bias/gelu/residual) ----
// Requires: M % 128 == 0, K % 32 == 0 (true for all call sites).
// ALIGNED_N: N % 128 == 0 -> fully unguarded vector staging.
#define BM 128
#define BN 128
#define BK 32
#define GF_BIAS 1
#define GF_GELU 2
#define GF_RES  4

template <bool ALIGNED_N>
__global__ __launch_bounds__(256)
void gemm_kernel(const float* __restrict__ A, const float* __restrict__ W,
                 const float* __restrict__ bias, const float* __restrict__ res,
                 float* __restrict__ C, int M, int N, int K, int flags) {
  __shared__ __align__(16) bf16 As[BM * BK];   // [m][k], k-contiguous
  __shared__ __align__(16) bf16 Bs[BN * BK];   // [n][k], k-contiguous

  const int tid  = threadIdx.x;
  const int lane = tid & 31;
  const int wid  = tid >> 5;        // 8 waves
  const int wm   = wid >> 1;        // 0..3 (4 x 32 rows)
  const int wn   = wid & 1;         // 0..1 (2 x 64 cols)
  const int g    = lane >> 4;       // half-wave
  const int hl   = lane & 15;
  const int m0   = blockIdx.y * BM;
  const int n0   = blockIdx.x * BN;

  v8f acc[2][4];
#pragma unroll
  for (int mt = 0; mt < 2; ++mt)
#pragma unroll
    for (int nt = 0; nt < 4; ++nt) acc[mt][nt] = zero8();

  for (int k0 = 0; k0 < K; k0 += BK) {
    // ---- stage A tile: 128x32 f32 -> bf16, float4 loads, packed b64 LDS stores
#pragma unroll
    for (int i = 0; i < 4; ++i) {
      int flat = tid + 256 * i;         // 0..1023 float4 chunks
      int m = flat >> 3;                // 8 chunks per 32-wide row
      int c = (flat & 7) * 4;
      float4 v = *(const float4*)&A[(size_t)(m0 + m) * K + k0 + c];
      *(v4bf*)&As[m * BK + c] = f2bf4(v);
    }
    // ---- stage B tile: 32x128 f32 -> bf16 LDS transposed to [n][k]
    if (ALIGNED_N) {
#pragma unroll
      for (int i = 0; i < 4; ++i) {
        int flat = tid + 256 * i;       // 0..1023 float4 chunks
        int k = flat >> 5;              // 32 chunks per 128-wide row
        int n = (flat & 31) * 4;
        float4 v = *(const float4*)&W[(size_t)(k0 + k) * N + n0 + n];
        Bs[(n + 0) * BK + k] = f2bf(v.x);
        Bs[(n + 1) * BK + k] = f2bf(v.y);
        Bs[(n + 2) * BK + k] = f2bf(v.z);
        Bs[(n + 3) * BK + k] = f2bf(v.w);
      }
    } else {
      // branchless edge handling: clamp address, select-zero out-of-range
#pragma unroll
      for (int i = 0; i < 16; ++i) {
        int flat = tid + 256 * i;
        int k = flat >> 7, n = flat & 127;
        int gn = n0 + n;
        int cn = gn < N ? gn : N - 1;
        float v = W[(size_t)(k0 + k) * N + cn];
        v = (gn < N) ? v : 0.f;
        Bs[n * BK + k] = f2bf(v);
      }
    }
    // ---- speculative prefetch of next K-tile into L2 (global_prefetch_b8)
    if (k0 + BK < K) {
      int pm = tid >> 1, pk = (tid & 1) * 16;
      __builtin_prefetch(&A[(size_t)(m0 + pm) * K + k0 + BK + pk], 0, 1);
      int bk2 = tid >> 3, bn2 = (tid & 7) * 16;
      __builtin_prefetch(&W[(size_t)(k0 + BK + bk2) * N + n0 + bn2], 0, 1);
    }
    __syncthreads();

    // ---- fragments (A row-per-lane K=16g+e, B col-per-lane K=16g+e)
    v16bf af[2], bfr[4];
#pragma unroll
    for (int mt = 0; mt < 2; ++mt) {
      const bf16* base = &As[(wm * 32 + mt * 16 + hl) * BK];
      af[mt] = cat16(*(const v8bf*)(base + 16 * g),
                     *(const v8bf*)(base + 16 * g + 8));
    }
#pragma unroll
    for (int nt = 0; nt < 4; ++nt) {
      const bf16* base = &Bs[(wn * 64 + nt * 16 + hl) * BK + 16 * g];
      bfr[nt] = cat16(*(const v8bf*)(base), *(const v8bf*)(base + 8));
    }
#pragma unroll
    for (int mt = 0; mt < 2; ++mt)
#pragma unroll
      for (int nt = 0; nt < 4; ++nt)
        acc[mt][nt] = wmma_bf16(af[mt], bfr[nt], acc[mt][nt]);
    __syncthreads();
  }

  // ---- epilogue: C layout — vgpr j holds row (8g + j), col = hl
#pragma unroll
  for (int mt = 0; mt < 2; ++mt) {
#pragma unroll
    for (int nt = 0; nt < 4; ++nt) {
      int col = n0 + wn * 64 + nt * 16 + hl;
      if (ALIGNED_N || col < N) {
        float bv = (flags & GF_BIAS) ? bias[col] : 0.f;
#pragma unroll
        for (int j = 0; j < 8; ++j) {
          int row = m0 + wm * 32 + mt * 16 + 8 * g + j;
          float v = acc[mt][nt][j] + bv;
          if (flags & GF_GELU) v = 0.5f * v * (1.f + erff(v * 0.70710678118654752f));
          if (flags & GF_RES)  v += res[(size_t)row * N + col];
          C[(size_t)row * N + col] = v;
        }
      }
    }
  }
}

// ---------------- fused causal flash attention, one wave per (b,h,16-q tile) ----
__global__ __launch_bounds__(32)
void attn_kernel(const float* __restrict__ qkv, float* __restrict__ y) {
  const int qb   = blockIdx.x;           // 0..T/16-1
  const int h    = blockIdx.y;
  const int b    = blockIdx.z;
  const int lane = threadIdx.x;
  const int g    = lane >> 4, hl = lane & 15;
  const int q0   = qb * 16;

  __shared__ __align__(16) bf16 Qs[16 * 64];   // [q][d]  (pre-scaled by 1/8)
  __shared__ __align__(16) bf16 Ks[32 * 64];   // [key][d]
  __shared__ __align__(16) bf16 Vs[64 * 32];   // [d][key] (transposed for B-frag)
  __shared__ __align__(16) bf16 Ps[16 * 32];   // [q][key]
  __shared__ float row_m[16], row_l[16];

  const size_t hoff = (size_t)h * DHEAD;

  // load + scale Q tile (16x64), float4 loads
#pragma unroll
  for (int i = 0; i < 8; ++i) {
    int flat = lane + 32 * i;          // 0..255 float4 chunks
    int r = flat >> 4;
    int c = (flat & 15) * 4;
    float4 v = *(const float4*)&qkv[(size_t)(b * T_SEQ + q0 + r) * D3 + hoff + c];
    v4bf q4;
    q4[0] = f2bf(v.x * 0.125f); q4[1] = f2bf(v.y * 0.125f);
    q4[2] = f2bf(v.z * 0.125f); q4[3] = f2bf(v.w * 0.125f);
    *(v4bf*)&Qs[r * 64 + c] = q4;
  }
  if (lane < 16) { row_m[lane] = -1e30f; row_l[lane] = 0.f; }
  asm volatile("s_wait_dscnt 0x0" ::: "memory");

  // Q fragments (loop invariant): step s covers feats s*32..s*32+31
  v16bf aq[2];
#pragma unroll
  for (int s = 0; s < 2; ++s) {
    const bf16* base = &Qs[hl * 64 + s * 32];
    aq[s] = cat16(*(const v8bf*)(base + 16 * g),
                  *(const v8bf*)(base + 16 * g + 8));
  }

  v8f acc[4];
#pragma unroll
  for (int o = 0; o < 4; ++o) acc[o] = zero8();

  const int tmax = q0 + 15;
  for (int kt = 0; kt <= tmax; kt += 32) {
    // stage K/V tiles (32 keys x 64 feats each). kt+31 <= 1023 < T always: no guard.
#pragma unroll 4
    for (int i = 0; i < 16; ++i) {
      int flat = lane + 32 * i;        // 0..511 float4 chunks
      int kk = flat >> 4;
      int c  = (flat & 15) * 4;
      const float* p = &qkv[(size_t)(b * T_SEQ + kt + kk) * D3 + hoff + c];
      float4 kv = *(const float4*)(p + D_EMB);
      float4 vv = *(const float4*)(p + 2 * D_EMB);
      *(v4bf*)&Ks[kk * 64 + c] = f2bf4(kv);
      Vs[(c + 0) * 32 + kk] = f2bf(vv.x);
      Vs[(c + 1) * 32 + kk] = f2bf(vv.y);
      Vs[(c + 2) * 32 + kk] = f2bf(vv.z);
      Vs[(c + 3) * 32 + kk] = f2bf(vv.w);
    }
    asm volatile("s_wait_dscnt 0x0" ::: "memory");

    // S = Q * K^T  -> two 16x16 tiles (keys kt..kt+15, kt+16..kt+31)
    v8f s0 = zero8(), s1 = zero8();
#pragma unroll
    for (int nt = 0; nt < 2; ++nt) {
      const bf16* base = &Ks[(nt * 16 + hl) * 64 + 16 * g];
      v16bf b0 = cat16(*(const v8bf*)(base),      *(const v8bf*)(base + 8));
      v16bf b1 = cat16(*(const v8bf*)(base + 32), *(const v8bf*)(base + 40));
      v8f s = zero8();
      s = wmma_bf16(aq[0], b0, s);
      s = wmma_bf16(aq[1], b1, s);
      if (nt == 0) s0 = s; else s1 = s;
    }

    // online softmax (row 8g+j spread across the 16 lanes of this half-wave)
    const int kc0 = kt + hl, kc1 = kt + 16 + hl;
#pragma unroll
    for (int j = 0; j < 8; ++j) {
      int row  = 8 * g + j;
      int qrow = q0 + row;
      float a0 = (kc0 <= qrow) ? s0[j] : -1e30f;
      float a1 = (kc1 <= qrow) ? s1[j] : -1e30f;
      float mx = fmaxf(a0, a1);
#pragma unroll
      for (int off = 1; off < 16; off <<= 1) mx = fmaxf(mx, __shfl_xor(mx, off, 32));
      float mo = row_m[row];
      float mn = fmaxf(mo, mx);
      float p0 = __expf(a0 - mn);
      float p1 = __expf(a1 - mn);
      float ps = p0 + p1;
#pragma unroll
      for (int off = 1; off < 16; off <<= 1) ps += __shfl_xor(ps, off, 32);
      float sc = __expf(mo - mn);
      float lnew = row_l[row] * sc + ps;
      if (hl == 0) { row_m[row] = mn; row_l[row] = lnew; }
#pragma unroll
      for (int o = 0; o < 4; ++o) acc[o][j] *= sc;   // rescale running output
      Ps[row * 32 + hl]      = f2bf(p0);
      Ps[row * 32 + 16 + hl] = f2bf(p1);
    }
    asm volatile("s_wait_dscnt 0x0" ::: "memory");

    // O += P * V  (P: 16x32, V: 32x64 -> 4 N-tiles)
    v16bf ap;
    {
      const bf16* base = &Ps[hl * 32];
      ap = cat16(*(const v8bf*)(base + 16 * g),
                 *(const v8bf*)(base + 16 * g + 8));
    }
#pragma unroll
    for (int o = 0; o < 4; ++o) {
      const bf16* vb = &Vs[(o * 16 + hl) * 32 + 16 * g];
      v16bf bv = cat16(*(const v8bf*)(vb), *(const v8bf*)(vb + 8));
      acc[o] = wmma_bf16(ap, bv, acc[o]);
    }
  }

  // normalize and write y[b,t,h*64+f]
#pragma unroll
  for (int j = 0; j < 8; ++j) {
    int row = 8 * g + j;
    float inv = 1.f / row_l[row];
#pragma unroll
    for (int o = 0; o < 4; ++o) {
      y[(size_t)(b * T_SEQ + q0 + row) * D_EMB + hoff + o * 16 + hl] =
          acc[o][j] * inv;
    }
  }
}

// ---------------- host orchestration ----------------
extern "C" void kernel_launch(void* const* d_in, const int* in_sizes, int n_in,
                              void* d_out, int out_size, void* d_ws, size_t ws_size,
                              hipStream_t stream) {
  (void)in_sizes; (void)n_in; (void)out_size; (void)ws_size;

  const int*   idx     = (const int*)  d_in[0];
  const float* tok_emb = (const float*)d_in[1];
  const float* pos_emb = (const float*)d_in[2];
  const float* ln1_w   = (const float*)d_in[3];
  const float* ln1_b   = (const float*)d_in[4];
  const float* attn_w  = (const float*)d_in[5];
  const float* attn_b  = (const float*)d_in[6];
  const float* proj_w  = (const float*)d_in[7];
  const float* proj_b  = (const float*)d_in[8];
  const float* ln2_w   = (const float*)d_in[9];
  const float* ln2_b   = (const float*)d_in[10];
  const float* fc_w    = (const float*)d_in[11];
  const float* fc_b    = (const float*)d_in[12];
  const float* fc2_w   = (const float*)d_in[13];
  const float* fc2_b   = (const float*)d_in[14];
  const float* lnf_w   = (const float*)d_in[15];
  const float* lnf_b   = (const float*)d_in[16];
  const float* head_w  = (const float*)d_in[17];
  const float* head_b  = (const float*)d_in[18];
  float* out = (float*)d_out;

  char* ws = (char*)d_ws;
  float* x    = (float*)ws; ws += (size_t)BT * D_EMB * sizeof(float);
  float* hbuf = (float*)ws; ws += (size_t)BT * D_EMB * sizeof(float);
  float* qkv  = (float*)ws; ws += (size_t)BT * D3    * sizeof(float);
  float* ybuf = (float*)ws; ws += (size_t)BT * D_EMB * sizeof(float);
  float* a4   = (float*)ws; ws += (size_t)BT * F4DIM * sizeof(float);

  const dim3 blk(256);
  auto ggrid = [](int M, int N) { return dim3((N + BN - 1) / BN, (M + BM - 1) / BM); };

  embed_kernel<<<BT, blk, 0, stream>>>(idx, tok_emb, pos_emb, x);

  for (int l = 0; l < N_LAYER; ++l) {
    ln_kernel<<<BT, blk, 0, stream>>>(x, ln1_w + (size_t)l * D_EMB,
                                      ln1_b + (size_t)l * D_EMB, hbuf);
    gemm_kernel<true><<<ggrid(BT, D3), blk, 0, stream>>>(
        hbuf, attn_w + (size_t)l * D_EMB * D3, attn_b + (size_t)l * D3,
        nullptr, qkv, BT, D3, D_EMB, GF_BIAS);
    attn_kernel<<<dim3(T_SEQ / 16, N_HEAD, B_BATCH), dim3(32), 0, stream>>>(qkv, ybuf);
    gemm_kernel<true><<<ggrid(BT, D_EMB), blk, 0, stream>>>(
        ybuf, proj_w + (size_t)l * D_EMB * D_EMB, proj_b + (size_t)l * D_EMB,
        x, x, BT, D_EMB, D_EMB, GF_BIAS | GF_RES);
    ln_kernel<<<BT, blk, 0, stream>>>(x, ln2_w + (size_t)l * D_EMB,
                                      ln2_b + (size_t)l * D_EMB, hbuf);
    gemm_kernel<true><<<ggrid(BT, F4DIM), blk, 0, stream>>>(
        hbuf, fc_w + (size_t)l * D_EMB * F4DIM, fc_b + (size_t)l * F4DIM,
        nullptr, a4, BT, F4DIM, D_EMB, GF_BIAS | GF_GELU);
    gemm_kernel<true><<<ggrid(BT, D_EMB), blk, 0, stream>>>(
        a4, fc2_w + (size_t)l * F4DIM * D_EMB, fc2_b + (size_t)l * D_EMB,
        x, x, BT, D_EMB, F4DIM, GF_BIAS | GF_RES);
  }

  ln_kernel<<<BT, blk, 0, stream>>>(x, lnf_w, lnf_b, hbuf);
  gemm_kernel<false><<<ggrid(BT, VOCAB), blk, 0, stream>>>(
      hbuf, head_w, head_b, nullptr, out, BT, VOCAB, D_EMB, GF_BIAS);
}
